// GraNNy_ViPeR_23210003268307
// MI455X (gfx1250) — matrix operand order repaired and checked by reading, not compiled, not run on hardware
//
#include <hip/hip_runtime.h>
#include <math.h>

typedef __attribute__((ext_vector_type(16))) __bf16 v16bf;
typedef __attribute__((ext_vector_type(8)))  float  v8f;

#define FDIM   128
#define NEDGE  1600000

struct U8 { unsigned int u[8]; };

__device__ __forceinline__ unsigned short f2bf(float f) {
  unsigned int u = __float_as_uint(f);
  u += 0x7FFFu + ((u >> 16) & 1u);           // round-to-nearest-even
  return (unsigned short)(u >> 16);
}

__device__ __forceinline__ void atomicMaxF(float* addr, float v) {
  if (v >= 0.f) atomicMax((int*)addr, __float_as_int(v));
  else          atomicMin((unsigned int*)addr, __float_as_uint(v));
}

// ---------------------------------------------------------------------------
// WMMA GEMM: C[M,128] = relu(A[M,K](bf16) * W[128,K](bf16)^T + bias)
// block = 256 thr (8 waves); wave w -> rows m0..m0+15, all 128 cols (8 tiles)
// W is staged once per block into LDS via async-to-LDS copies (CDNA5 path),
// with +8-half row padding so 16 row-parallel lane reads hit distinct banks.
// ---------------------------------------------------------------------------
template <int K>
__global__ void __launch_bounds__(256)
gemm_bf16_relu(const unsigned short* __restrict__ A, int lda,
               const unsigned short* __restrict__ W,
               const float* __restrict__ bias,
               float* __restrict__ C)
{
  extern __shared__ unsigned short ldsW[];
  constexpr int LDW = K + 8;                    // padded LDS row stride (halves)

  // --- cooperative async copy of W[128,K] into LDS ------------------------
  {
    constexpr int CPR = K >> 3;                 // 16B chunks per row
    constexpr int NCH = 128 * CPR;              // multiple of 256 -> full EXEC
#pragma unroll
    for (int c = threadIdx.x; c < NCH; c += 256) {
      int row = c / CPR, col = (c % CPR) * 8;
      const unsigned short* g = W + (size_t)row * K + col;
      unsigned int l = (unsigned int)(size_t)(ldsW + row * LDW + col);
      asm volatile("global_load_async_to_lds_b128 %0, %1, off"
                   :: "v"(l), "v"(g) : "memory");
    }
    asm volatile("s_wait_asynccnt 0" ::: "memory");
    __syncthreads();
  }

  const int wave = threadIdx.x >> 5;
  const int lane = threadIdx.x & 31;
  const int half = lane >> 4;
  const int l16  = lane & 15;
  const int m0   = blockIdx.x * 128 + wave * 16;
  const unsigned short* arow = A + (size_t)(m0 + l16) * lda;

  v8f acc[8];
#pragma unroll
  for (int t = 0; t < 8; ++t) acc[t] = (v8f){0.f,0.f,0.f,0.f,0.f,0.f,0.f,0.f};

#pragma unroll
  for (int k0 = 0; k0 < K; k0 += 32) {
    U8 au;
#pragma unroll
    for (int v = 0; v < 8; ++v) {              // A 16x32 bf16 lane layout
      int kk = k0 + ((v < 4) ? (2 * v) : (16 + 2 * (v - 4))) + half * 8;
      au.u[v] = *(const unsigned int*)(arow + kk);
    }
    v16bf av = __builtin_bit_cast(v16bf, au);
#pragma unroll
    for (int t = 0; t < 8; ++t) {
      U8 bu;
      const unsigned short* wrow = ldsW + (t * 16 + l16) * LDW;
#pragma unroll
      for (int v = 0; v < 8; ++v) {            // B 32x16 bf16 lane layout
        int kk = k0 + ((v < 4) ? (2 * v) : (16 + 2 * (v - 4))) + half * 8;
        bu.u[v] = *(const unsigned int*)(wrow + kk);
      }
      v16bf bv = __builtin_bit_cast(v16bf, bu);
      acc[t] = __builtin_amdgcn_wmma_f32_16x16x32_bf16(
                 false, av, false, bv, (short)0, acc[t], false, false);
    }
  }

#pragma unroll
  for (int t = 0; t < 8; ++t) {
    int n = t * 16 + l16;
    float bv = bias ? bias[n] : 0.f;
#pragma unroll
    for (int v = 0; v < 8; ++v) {
      int m = m0 + v + 8 * half;               // C/D layout: m = vgpr + 8*half
      float o = acc[t][v] + bv;
      C[(size_t)m * FDIM + n] = o > 0.f ? o : 0.f;
    }
  }
}

// ---------------------------------------------------------------------------
// f32 [rows,128] -> bf16 into abf[rows,256] at column offset; zero-pad rows
// ---------------------------------------------------------------------------
__global__ void cvt_to_abf(const float* __restrict__ src,
                           unsigned short* __restrict__ abf,
                           int colOff, int Msrc, int Mpad)
{
  int t = blockIdx.x * 256 + threadIdx.x;
  int row = t >> 5, c0 = (t & 31) * 4;
  if (row >= Mpad) return;
  float4 v = {0.f, 0.f, 0.f, 0.f};
  if (row < Msrc) v = *(const float4*)(src + (size_t)row * FDIM + c0);
  unsigned int p0 = (unsigned int)f2bf(v.x) | ((unsigned int)f2bf(v.y) << 16);
  unsigned int p1 = (unsigned int)f2bf(v.z) | ((unsigned int)f2bf(v.w) << 16);
  unsigned int* d = (unsigned int*)(abf + (size_t)row * 256 + colOff + c0);
  d[0] = p0; d[1] = p1;
}

__global__ void cvt_w(const float* __restrict__ s, unsigned short* __restrict__ d, int n)
{
  int t = blockIdx.x * 256 + threadIdx.x;
  if (t < n) d[t] = f2bf(s[t]);
}

__global__ void copy_f4(const float* __restrict__ s, float* __restrict__ d, int n4)
{
  int t = blockIdx.x * 256 + threadIdx.x;
  if (t < n4) ((float4*)d)[t] = ((const float4*)s)[t];
}

// ---------------------------------------------------------------------------
// Edge scatter-max: aggr[dst] = max(aggr[dst], y[src]); y >= 0 -> int atomics
// ---------------------------------------------------------------------------
__global__ void scatter_max(const int* __restrict__ edges,
                            const float* __restrict__ y,
                            float* __restrict__ aggr)
{
  int t = blockIdx.x * 256 + threadIdx.x;
  int e = t >> 5;
  if (e >= NEDGE) return;
  int f = (t & 31) * 4;
  int s = edges[e], d = edges[NEDGE + e];
  float4 v = *(const float4*)(y + (size_t)s * FDIM + f);
  int* a = (int*)(aggr + (size_t)d * FDIM + f);
  atomicMax(a + 0, __float_as_int(v.x));
  atomicMax(a + 1, __float_as_int(v.y));
  atomicMax(a + 2, __float_as_int(v.z));
  atomicMax(a + 3, __float_as_int(v.w));
}

// ---------------------------------------------------------------------------
// TopK pooling pieces
// ---------------------------------------------------------------------------
__global__ void wnorm_k(const float* __restrict__ wp, float* __restrict__ invn)
{
  __shared__ float sh[128];
  int t = threadIdx.x;
  float v = wp[t];
  sh[t] = v * v;
  __syncthreads();
  for (int o = 64; o > 0; o >>= 1) { if (t < o) sh[t] += sh[t + o]; __syncthreads(); }
  if (t == 0) invn[0] = rsqrtf(sh[0]);
}

__global__ void node_scores(const float* __restrict__ h, const float* __restrict__ wp,
                            const float* __restrict__ invn,
                            float* __restrict__ s, unsigned int* __restrict__ keys, int M)
{
  int t = blockIdx.x * 256 + threadIdx.x;
  int nod = t >> 5, lane = t & 31;
  if (nod >= M) return;
  float4 xv = *(const float4*)(h + (size_t)nod * FDIM + lane * 4);
  float4 wv = *(const float4*)(wp + lane * 4);
  float d = xv.x * wv.x + xv.y * wv.y + xv.z * wv.z + xv.w * wv.w;
#pragma unroll
  for (int o = 16; o > 0; o >>= 1) d += __shfl_xor(d, o, 32);
  if (lane == 0) {
    float sc = d * invn[0];
    s[nod] = sc;
    unsigned int u = __float_as_uint(sc);
    keys[nod] = (u & 0x80000000u) ? ~u : (u | 0x80000000u);  // order-preserving flip
  }
}

// per-graph radix-select of k-th largest key; also resets compaction counters
__global__ void select_kth(const unsigned int* __restrict__ keys, int n, int k,
                           unsigned int* __restrict__ selinfo, int* __restrict__ ctr)
{
  int g = blockIdx.x;
  const unsigned int* kg = keys + (size_t)g * n;
  __shared__ int cnt;
  unsigned int prefix = 0;
  for (int b = 31; b >= 0; --b) {
    unsigned int tv = prefix | (1u << b);
    if (threadIdx.x == 0) cnt = 0;
    __syncthreads();
    int c = 0;
    for (int i = threadIdx.x; i < n; i += blockDim.x) c += (kg[i] >= tv) ? 1 : 0;
    if (c) atomicAdd(&cnt, c);
    __syncthreads();
    if (cnt >= k) prefix = tv;
    __syncthreads();
  }
  if (threadIdx.x == 0) cnt = 0;
  __syncthreads();
  int c = 0;
  for (int i = threadIdx.x; i < n; i += blockDim.x) c += (kg[i] > prefix) ? 1 : 0;
  if (c) atomicAdd(&cnt, c);
  __syncthreads();
  if (threadIdx.x == 0) {
    selinfo[2 * g] = prefix;
    selinfo[2 * g + 1] = (unsigned int)cnt;
    ctr[2 * g] = 0; ctr[2 * g + 1] = 0;
  }
}

__global__ void compact_topk(const unsigned int* __restrict__ keys,
                             const float* __restrict__ scores, int n, int k,
                             const unsigned int* __restrict__ selinfo,
                             int* __restrict__ ctr,
                             int* __restrict__ gidx, float* __restrict__ vals)
{
  int g = blockIdx.y;
  int i = blockIdx.x * 256 + threadIdx.x;
  if (i >= n) return;
  int node = g * n + i;
  unsigned int key = keys[node];
  unsigned int T = selinfo[2 * g];
  int gt = (int)selinfo[2 * g + 1];
  int pos = -1;
  if (key > T) pos = atomicAdd(&ctr[2 * g], 1);
  else if (key == T) {
    int q = atomicAdd(&ctr[2 * g + 1], 1);
    if (q < k - gt) pos = gt + q;
  }
  if (pos >= 0 && pos < k) {
    gidx[g * k + pos] = node;
    vals[g * k + pos] = scores[node];
  }
}

__global__ void pool_gather(const float* __restrict__ h, const int* __restrict__ gidx,
                            const float* __restrict__ vals, float* __restrict__ xnew, int Mn)
{
  int t = blockIdx.x * 256 + threadIdx.x;
  int p = t >> 5;
  if (p >= Mn) return;
  int f = (t & 31) * 4;
  float sc = tanhf(vals[p]);
  float4 v = *(const float4*)(h + (size_t)gidx[p] * FDIM + f);
  float4 o = {v.x * sc, v.y * sc, v.z * sc, v.w * sc};
  *(float4*)(xnew + (size_t)p * FDIM + f) = o;
}

__global__ void newpos_init(int* __restrict__ np, int n, int Mn)
{
  int t = blockIdx.x * 256 + threadIdx.x;
  if (t < n) np[t] = Mn;
}

__global__ void newpos_scatter(int* __restrict__ np, const int* __restrict__ gidx, int Mn)
{
  int t = blockIdx.x * 256 + threadIdx.x;
  if (t < Mn) np[gidx[t]] = t;
}

__global__ void remap_edges(const int* __restrict__ ein, int* __restrict__ eout,
                            const int* __restrict__ np, int Mn)
{
  int e = blockIdx.x * 256 + threadIdx.x;
  if (e >= NEDGE) return;
  int ns = np[ein[e]];
  int nd = np[ein[NEDGE + e]];
  if (ns == Mn || nd == Mn) { ns = Mn; nd = Mn; }
  eout[e] = ns; eout[NEDGE + e] = nd;
}

// ---------------------------------------------------------------------------
// Readout: per-graph max (cols 0..127) and mean (cols 128..255) accumulated
// ---------------------------------------------------------------------------
__global__ void readout_partial(const float* __restrict__ x, int kn,
                                float* __restrict__ rmax, float* __restrict__ rsum)
{
  int g = blockIdx.y, chunk = blockIdx.x, f = threadIdx.x;
  int per = (kn + 15) / 16;
  int i0 = chunk * per, i1 = min(kn, i0 + per);
  float mx = -INFINITY, sm = 0.f;
  for (int i = i0; i < i1; ++i) {
    float v = x[(size_t)(g * kn + i) * FDIM + f];
    mx = fmaxf(mx, v); sm += v;
  }
  atomicMaxF(&rmax[g * FDIM + f], mx);
  atomicAdd(&rsum[g * FDIM + f], sm);
}

__global__ void readout_fin(float* __restrict__ z, float* __restrict__ rmax,
                            float* __restrict__ rsum, float invk)
{
  int t = blockIdx.x * 256 + threadIdx.x;
  if (t >= 1024) return;
  int g = t >> 7, f = t & 127;
  z[g * 256 + f]       += rmax[t];
  z[g * 256 + 128 + f] += rsum[t] * invk;
  rmax[t] = -INFINITY; rsum[t] = 0.f;
}

__global__ void init_all(float* __restrict__ z, float* __restrict__ rmax,
                         float* __restrict__ rsum)
{
  int t = blockIdx.x * 256 + threadIdx.x;
  if (t < 2048) z[t] = 0.f;
  if (t < 1024) { rmax[t] = -INFINITY; rsum[t] = 0.f; }
}

// ---------------------------------------------------------------------------
// MLP head: z[8,256] -> relu(Wl1) -> relu(Wl2) -> sigmoid(Wl3) -> out[8]
// ---------------------------------------------------------------------------
__global__ void __launch_bounds__(1024)
mlp_head(const float* __restrict__ z,
         const float* __restrict__ Wl1, const float* __restrict__ bl1,
         const float* __restrict__ Wl2, const float* __restrict__ bl2,
         const float* __restrict__ Wl3, const float* __restrict__ bl3,
         float* __restrict__ out)
{
  __shared__ float t1[8 * 128];
  __shared__ float t2[8 * 64];
  int tid = threadIdx.x;
  { int b = tid >> 7, j = tid & 127;
    float a = bl1[j];
    for (int k = 0; k < 256; ++k) a += z[b * 256 + k] * Wl1[j * 256 + k];
    t1[b * 128 + j] = a > 0.f ? a : 0.f; }
  __syncthreads();
  if (tid < 512) {
    int b = tid >> 6, j = tid & 63;
    float a = bl2[j];
    for (int k = 0; k < 128; ++k) a += t1[b * 128 + k] * Wl2[j * 128 + k];
    t2[b * 64 + j] = a > 0.f ? a : 0.f;
  }
  __syncthreads();
  if (tid < 8) {
    float a = bl3[0];
    for (int k = 0; k < 64; ++k) a += t2[tid * 64 + k] * Wl3[k];
    out[tid] = 1.f / (1.f + expf(-a));
  }
}

// ---------------------------------------------------------------------------
extern "C" void kernel_launch(void* const* d_in, const int* in_sizes, int n_in,
                              void* d_out, int out_size, void* d_ws, size_t ws_size,
                              hipStream_t stream)
{
  (void)in_sizes; (void)n_in; (void)out_size; (void)ws_size;
  const float* x0      = (const float*)d_in[0];
  const int*   edges0  = (const int*)d_in[1];
  const float* Wlin[3] = {(const float*)d_in[2], (const float*)d_in[6], (const float*)d_in[10]};
  const float* blin[3] = {(const float*)d_in[3], (const float*)d_in[7], (const float*)d_in[11]};
  const float* Wupd[3] = {(const float*)d_in[4], (const float*)d_in[8], (const float*)d_in[12]};
  const float* wpv[3]  = {(const float*)d_in[5], (const float*)d_in[9], (const float*)d_in[13]};
  const float* Wl1 = (const float*)d_in[14]; const float* bl1 = (const float*)d_in[15];
  const float* Wl2 = (const float*)d_in[16]; const float* bl2 = (const float*)d_in[17];
  const float* Wl3 = (const float*)d_in[18]; const float* bl3 = (const float*)d_in[19];

  const int MPADMAX = 100096;
  char* base = (char*)d_ws;
  size_t off = 0;
  auto take = [&](size_t bytes) -> void* {
    void* p = base + off; off = (off + bytes + 255) & ~(size_t)255; return p;
  };
  unsigned short* abf  = (unsigned short*)take((size_t)MPADMAX * 256 * 2);
  float* ybuf  = (float*)take((size_t)MPADMAX * 128 * 4);
  float* aggr  = (float*)take((size_t)MPADMAX * 128 * 4);
  float* xcur  = (float*)take((size_t)MPADMAX * 128 * 4);
  int*   edg   = (int*)take((size_t)2 * NEDGE * 4);
  unsigned short* wlbf[3]; unsigned short* wubf[3];
  for (int i = 0; i < 3; ++i) wlbf[i] = (unsigned short*)take(128 * 128 * 2);
  for (int i = 0; i < 3; ++i) wubf[i] = (unsigned short*)take(128 * 256 * 2);
  float* sbuf = (float*)take(100000 * 4);
  unsigned int* keys = (unsigned int*)take(100000 * 4);
  int*   gidx = (int*)take(80000 * 4);
  float* vals = (float*)take(80000 * 4);
  int*   npos = (int*)take(100016 * 4);
  unsigned int* selinfo = (unsigned int*)take(64);
  int*   ctr  = (int*)take(64);
  float* z    = (float*)take(2048 * 4);
  float* rmax = (float*)take(1024 * 4);
  float* rsum = (float*)take(1024 * 4);
  float* invn = (float*)take(256);

  auto G = [](long n) { return (unsigned int)((n + 255) / 256); };

  // weight conversion + accumulator init
  for (int i = 0; i < 3; ++i) {
    cvt_w<<<G(128 * 128), 256, 0, stream>>>(Wlin[i], wlbf[i], 128 * 128);
    cvt_w<<<G(128 * 256), 256, 0, stream>>>(Wupd[i], wubf[i], 128 * 256);
  }
  init_all<<<8, 256, 0, stream>>>(z, rmax, rsum);

  const int Ms[4] = {100000, 80000, 64000, 51200};
  const int ns[3] = {12500, 10000, 8000};
  const int ks[3] = {10000, 8000, 6400};
  const float* xin = x0;
  const int*   ein = edges0;

  const size_t sh128 = (size_t)128 * (128 + 8) * 2;   // 34,816 B
  const size_t sh256 = (size_t)128 * (256 + 8) * 2;   // 67,584 B

  for (int L = 0; L < 3; ++L) {
    const int M = Ms[L], n = ns[L], k = ks[L], Mn = Ms[L + 1];
    const int Mpad = ((M + 1 + 127) / 128) * 128;

    // SAGE conv: y = relu(x_ext @ Wlin^T + b)
    cvt_to_abf<<<G((long)Mpad * 32), 256, 0, stream>>>(xin, abf, 128, M, Mpad);
    gemm_bf16_relu<128><<<Mpad / 128, 256, sh128, stream>>>(abf + 128, 256, wlbf[L], blin[L], ybuf);
    // aggr init (self loops) + edge scatter-max
    copy_f4<<<G((long)Mpad * 32), 256, 0, stream>>>(ybuf, aggr, Mpad * 32);
    scatter_max<<<G((long)NEDGE * 32), 256, 0, stream>>>(ein, ybuf, aggr);
    // h = relu([aggr || x_ext] @ Wupd^T)
    cvt_to_abf<<<G((long)Mpad * 32), 256, 0, stream>>>(aggr, abf, 0, M + 1, Mpad);
    gemm_bf16_relu<256><<<Mpad / 128, 256, sh256, stream>>>(abf, 256, wubf[L], nullptr, ybuf);

    // TopK pooling
    wnorm_k<<<1, 128, 0, stream>>>(wpv[L], invn);
    node_scores<<<G((long)M * 32), 256, 0, stream>>>(ybuf, wpv[L], invn, sbuf, keys, M);
    select_kth<<<8, 256, 0, stream>>>(keys, n, k, selinfo, ctr);
    dim3 cg(G(n), 8);
    compact_topk<<<cg, 256, 0, stream>>>(keys, sbuf, n, k, selinfo, ctr, gidx, vals);
    pool_gather<<<G((long)Mn * 32), 256, 0, stream>>>(ybuf, gidx, vals, xcur, Mn);
    newpos_init<<<G(M + 1), 256, 0, stream>>>(npos, M + 1, Mn);
    newpos_scatter<<<G(Mn), 256, 0, stream>>>(npos, gidx, Mn);
    remap_edges<<<G(NEDGE), 256, 0, stream>>>(ein, edg, npos, Mn);

    // readout accumulate (z += [gmp || gap])
    dim3 rg(16, 8);
    readout_partial<<<rg, 128, 0, stream>>>(xcur, k, rmax, rsum);
    readout_fin<<<4, 256, 0, stream>>>(z, rmax, rsum, 1.f / (float)k);

    xin = xcur; ein = edg;
  }

  mlp_head<<<1, 1024, 0, stream>>>(z, Wl1, bl1, Wl2, bl2, Wl3, bl3, (float*)d_out);
}
